// MambaBlock_66142496359002
// MI455X (gfx1250) — matrix operand order repaired and checked by reading, not compiled, not run on hardware
//
#include <hip/hip_runtime.h>
#include <hip/hip_bf16.h>
#include <math.h>

// ---------------------------------------------------------------------------
// Mamba block for MI455X (gfx1250, wave32, WMMA, async-to-LDS).
//   B=2, L=2048, d_model=1024, d_inner=2048, d_state=16, dt_rank=64, d_conv=4
//
// Strategy: all GEMM math is v_wmma_f32_16x16x32_f16, so keep every GEMM
// operand / intermediate activation in f16 (fp32->f16 pre-pass for x and the
// 4 weight matrices). GEMM tiles are staged into LDS with
// GLOBAL_LOAD_ASYNC_TO_LDS_B128 + s_wait_asynccnt double buffering when the
// toolchain exposes the builtins (sync uint4 staging fallback otherwise).
// ---------------------------------------------------------------------------

typedef __attribute__((ext_vector_type(16))) _Float16 v16h;
typedef __attribute__((ext_vector_type(8)))  float    v8f;
typedef int v4i __attribute__((vector_size(16)));    // builtin's b128 unit

union FragH { v16h h; uint4 q[2]; };
union AccF  { v8f  v; float f[8]; };

__device__ __forceinline__ float softplus_f(float x) {
    return (x > 20.0f) ? x : log1pf(__expf(x));
}
__device__ __forceinline__ float silu_f(float x) {
    return x / (1.0f + __expf(-x));
}

// ---- gfx1250 async global->LDS copy (ASYNCcnt) with portable fallback ------
#if defined(__gfx1250__) && __has_builtin(__builtin_amdgcn_global_load_async_to_lds_b128)
#define HAVE_ASYNC_LDS 1
#endif

#if defined(HAVE_ASYNC_LDS)
__device__ __forceinline__ void async_copy16(const void* g, void* l) {
    // generic->AS pointers via integer round-trip: global generic addr == AS1
    // addr; LDS generic addr low 32 bits == LDS byte offset (flat aperture).
    __builtin_amdgcn_global_load_async_to_lds_b128(
        (__attribute__((address_space(1))) v4i*)(uintptr_t)g,
        (__attribute__((address_space(3))) v4i*)(uint32_t)(uintptr_t)l,
        0, 0);
}
template <int N>
__device__ __forceinline__ void wait_async() {
#if __has_builtin(__builtin_amdgcn_s_wait_asynccnt)
    __builtin_amdgcn_s_wait_asynccnt(N);
#else
    asm volatile("s_wait_asynccnt %0" ::"i"(N));
#endif
}
#endif

// ---------------------------------------------------------------------------
// fp32 -> f16 conversion pre-pass.
// ---------------------------------------------------------------------------
__global__ void cvt_f32_to_f16(const float* __restrict__ s,
                               _Float16* __restrict__ d, long n)
{
    const long i = (long)blockIdx.x * blockDim.x + threadIdx.x;
    if (i < n) d[i] = (_Float16)s[i];
}

// ---------------------------------------------------------------------------
// C[M,N] = X[M,K] * W[N,K]^T, all-f16 operands, f32 WMMA accumulation.
//   EPI  == 1 : fuse bias (fp32) + softplus into epilogue
//   OF16 == 1 : store C as f16, else fp32
// Block: 256 threads = 8 waves, block tile 64x64, K step 32, double-buffered
// LDS. Out-of-range A/B rows are clamped (not zeroed): the resulting garbage
// only reaches output rows >= M / columns >= N, which the store guards drop.
// ---------------------------------------------------------------------------
template<int EPI, int OF16>
__global__ __launch_bounds__(256) void gemm_f16_wmma(
    const _Float16* __restrict__ X, const _Float16* __restrict__ W,
    const float* __restrict__ bias, void* __restrict__ Cv,
    int M, int N, int K, int lda, int ldc)
{
    __shared__ __align__(16) _Float16 As[2][64 * 32];   // [m][k]
    __shared__ __align__(16) _Float16 Bs[2][64 * 32];   // [n][k] (== B^T)

    const int tid  = threadIdx.x;
    const int lane = tid & 31;
    const int wv   = tid >> 5;
    const int wm   = wv & 3;          // M sub-tile (0..3)
    const int wn   = wv >> 2;         // N half (0..1)
    const int lr   = lane & 15;
    const int ks   = lane >> 4;

    const int bm = blockIdx.y;
    const int bn = blockIdx.x;

    // cooperative copy mapping: thread -> (row 0..63, 8 consecutive k)
    const int lm = tid >> 2;
    const int kb = (tid & 3) * 8;
    const int gmL = bm * 64 + lm;
    const int gnL = bn * 64 + lm;
    const long arow = (long)(gmL < M ? gmL : M - 1) * lda;   // clamped
    const long brow = (long)(gnL < N ? gnL : N - 1) * K;     // clamped

    v8f acc0 = {}; v8f acc1 = {};

    auto compute = [&](const _Float16* Ab, const _Float16* Bb) {
        // A 16x32 fragment: lane lr = M row, elems -> K = ks*16 + 0..15.
        FragH a;
        const uint4* ar = (const uint4*)(Ab + (wm * 16 + lr) * 32);
        a.q[0] = ar[2 * ks];
        a.q[1] = ar[2 * ks + 1];
        // B 32x16 fragments: lane lr = N col, elems -> K = ks*16 + 0..15.
        FragH b0, b1;
        const uint4* br0 = (const uint4*)(Bb + (wn * 32 + lr) * 32);
        const uint4* br1 = (const uint4*)(Bb + (wn * 32 + 16 + lr) * 32);
        b0.q[0] = br0[2 * ks]; b0.q[1] = br0[2 * ks + 1];
        b1.q[0] = br1[2 * ks]; b1.q[1] = br1[2 * ks + 1];
        acc0 = __builtin_amdgcn_wmma_f32_16x16x32_f16(
                   false, a.h, false, b0.h, (short)0, acc0, false, false);
        acc1 = __builtin_amdgcn_wmma_f32_16x16x32_f16(
                   false, a.h, false, b1.h, (short)0, acc1, false, false);
    };

    const int nk = K >> 5;

#if defined(HAVE_ASYNC_LDS)
    auto stage = [&](int k0, int buf) {   // 2 async instructions per wave
        async_copy16(X + arow + k0 + kb, &As[buf][lm * 32 + kb]);
        async_copy16(W + brow + k0 + kb, &Bs[buf][lm * 32 + kb]);
    };
    stage(0, 0);
    for (int kk = 0; kk < nk; ++kk) {
        const int buf = kk & 1;
        if (kk + 1 < nk) {
            stage((kk + 1) << 5, buf ^ 1);  // buffer last read in previous iter
            wait_async<2>();                // retire current stage only
        } else {
            wait_async<0>();
        }
        __syncthreads();
        compute(As[buf], Bs[buf]);
        __syncthreads();                    // readers done before overwrite
    }
#else
    for (int kk = 0; kk < nk; ++kk) {
        const int k0 = kk << 5;
        const uint4 av = *(const uint4*)(X + arow + k0 + kb);
        const uint4 bv = *(const uint4*)(W + brow + k0 + kb);
        *(uint4*)&As[0][lm * 32 + kb] = av;
        *(uint4*)&Bs[0][lm * 32 + kb] = bv;
        __syncthreads();
        compute(As[0], Bs[0]);
        __syncthreads();
    }
#endif

    // ---- store C (16x16 f32 layout: m = ks*8 + i, n = lr) ----
    AccF r0, r1; r0.v = acc0; r1.v = acc1;
    const int gmS = bm * 64 + wm * 16 + ks * 8;
    const int gn0 = bn * 64 + wn * 32 + lr;
    const int gn1 = gn0 + 16;
    const float bb0 = (EPI == 1 && gn0 < N) ? bias[gn0] : 0.0f;
    const float bb1 = (EPI == 1 && gn1 < N) ? bias[gn1] : 0.0f;
    #pragma unroll
    for (int i = 0; i < 8; ++i) {
        const int m = gmS + i;
        if (m < M) {
            if (gn0 < N) {
                float v = r0.f[i];
                if (EPI == 1) v = softplus_f(v + bb0);
                if (OF16) ((_Float16*)Cv)[(long)m * ldc + gn0] = (_Float16)v;
                else      ((float*)Cv)[(long)m * ldc + gn0] = v;
            }
            if (gn1 < N) {
                float v = r1.f[i];
                if (EPI == 1) v = softplus_f(v + bb1);
                if (OF16) ((_Float16*)Cv)[(long)m * ldc + gn1] = (_Float16)v;
                else      ((float*)Cv)[(long)m * ldc + gn1] = v;
            }
        }
    }
}

// ---------------------------------------------------------------------------
// Causal depthwise conv1d (width 4) + SiLU over xz[:, :, 0:2048] (f16 I/O).
// ---------------------------------------------------------------------------
__global__ void conv_silu_kernel(const _Float16* __restrict__ xz,
                                 const float* __restrict__ w,
                                 const float* __restrict__ b,
                                 _Float16* __restrict__ xc,
                                 int BL, int L)
{
    const int idx = blockIdx.x * blockDim.x + threadIdx.x;
    if (idx >= BL * 2048) return;
    const int d   = idx & 2047;
    const int row = idx >> 11;        // b*L + l
    const int l   = row % L;
    float acc = b[d];
    #pragma unroll
    for (int j = 0; j < 4; ++j) {
        const int ll = l - 3 + j;
        if (ll >= 0)
            acc += w[d * 4 + j] * (float)xz[(long)(row - 3 + j) * 4096 + d];
    }
    xc[(long)row * 2048 + d] = (_Float16)silu_f(acc);
}

// ---------------------------------------------------------------------------
// Selective scan: one thread per (batch, channel); 16-state SSM in fp32 regs.
// ---------------------------------------------------------------------------
__global__ void scan_kernel(const _Float16* __restrict__ dt,
                            const _Float16* __restrict__ xdb,
                            const _Float16* __restrict__ xc,
                            const _Float16* __restrict__ xz,
                            const float* __restrict__ A_log,
                            const float* __restrict__ Dp,
                            _Float16* __restrict__ y,
                            int B, int L)
{
    const int idx = blockIdx.x * blockDim.x + threadIdx.x;
    if (idx >= B * 2048) return;
    const int b = idx >> 11;
    const int d = idx & 2047;

    float A[16], h[16];
    #pragma unroll
    for (int s = 0; s < 16; ++s) { A[s] = -__expf(A_log[d * 16 + s]); h[s] = 0.0f; }
    const float Dd = Dp[d];

    long row = (long)b * L;
    for (int l = 0; l < L; ++l, ++row) {
        const float dtv = (float)dt[row * 2048 + d];
        const float xcv = (float)xc[row * 2048 + d];
        const float zv  = (float)xz[row * 4096 + 2048 + d];
        const _Float16* bc = xdb + row * 96;   // [dt(64) | B(16) | C(16)]
        float yv = 0.0f;
        #pragma unroll
        for (int s = 0; s < 16; ++s) {
            const float dA = __expf(dtv * A[s]);
            h[s] = dA * h[s] + (dtv * (float)bc[64 + s]) * xcv;
            yv  += h[s] * (float)bc[80 + s];
        }
        yv += Dd * xcv;
        y[row * 2048 + d] = (_Float16)(yv * silu_f(zv));
    }
}

// ---------------------------------------------------------------------------
// Residual add + LayerNorm over 1024 columns (one block per row, fp32).
// ---------------------------------------------------------------------------
__global__ __launch_bounds__(256) void ln_kernel(
    const float* __restrict__ otmp, const float* __restrict__ x,
    const float* __restrict__ w, const float* __restrict__ b,
    float* __restrict__ out, int ncol)
{
    __shared__ float r1[256];
    __shared__ float r2[256];
    const int row = blockIdx.x;
    const float* ho = otmp + (long)row * ncol;
    const float* hx = x    + (long)row * ncol;

    float vals[4];
    float s = 0.0f, s2 = 0.0f;
    #pragma unroll
    for (int i = 0; i < 4; ++i) {
        const int c = threadIdx.x + i * 256;
        const float h = ho[c] + hx[c];
        vals[i] = h; s += h; s2 += h * h;
    }
    r1[threadIdx.x] = s; r2[threadIdx.x] = s2;
    __syncthreads();
    for (int off = 128; off > 0; off >>= 1) {
        if (threadIdx.x < off) {
            r1[threadIdx.x] += r1[threadIdx.x + off];
            r2[threadIdx.x] += r2[threadIdx.x + off];
        }
        __syncthreads();
    }
    const float mu  = r1[0] / (float)ncol;
    const float var = r2[0] / (float)ncol - mu * mu;
    const float inv = rsqrtf(var + 1e-5f);
    #pragma unroll
    for (int i = 0; i < 4; ++i) {
        const int c = threadIdx.x + i * 256;
        out[(long)row * ncol + c] = (vals[i] - mu) * inv * w[c] + b[c];
    }
}

// ---------------------------------------------------------------------------
extern "C" void kernel_launch(void* const* d_in, const int* in_sizes, int n_in,
                              void* d_out, int out_size, void* d_ws, size_t ws_size,
                              hipStream_t stream)
{
    const float* x          = (const float*)d_in[0];
    const float* in_proj_w  = (const float*)d_in[1];
    const float* conv_w     = (const float*)d_in[2];
    const float* conv_b     = (const float*)d_in[3];
    const float* x_proj_w   = (const float*)d_in[4];
    const float* dt_proj_w  = (const float*)d_in[5];
    const float* dt_proj_b  = (const float*)d_in[6];
    const float* A_log      = (const float*)d_in[7];
    const float* Dp         = (const float*)d_in[8];
    const float* out_proj_w = (const float*)d_in[9];
    const float* ln_w       = (const float*)d_in[10];
    const float* ln_b       = (const float*)d_in[11];
    float* out = (float*)d_out;

    const int B = 2, L = 2048, DM = 1024, DI = 2048;
    const int BL = B * L;                          // 4096

    // ---- workspace layout: f16 pool then fp32 otmp (~122 MB total) ----
    _Float16* hp   = (_Float16*)d_ws;
    _Float16* x16  = hp;                            // BL*DM      = 4194304
    _Float16* wi16 = x16  + (size_t)BL * DM;        // 2*DI*DM    = 4194304
    _Float16* wx16 = wi16 + (size_t)2 * DI * DM;    // 96*DI      = 196608
    _Float16* wd16 = wx16 + (size_t)96 * DI;        // DI*64      = 131072
    _Float16* wo16 = wd16 + (size_t)DI * 64;        // DM*DI      = 2097152
    _Float16* xz16 = wo16 + (size_t)DM * DI;        // BL*2*DI    = 16777216
    _Float16* xc16 = xz16 + (size_t)BL * 2 * DI;    // BL*DI      = 8388608
    _Float16* xdb16= xc16 + (size_t)BL * DI;        // BL*96      = 393216
    _Float16* dt16 = xdb16+ (size_t)BL * 96;        // BL*DI      = 8388608
    _Float16* y16  = dt16 + (size_t)BL * DI;        // BL*DI      = 8388608
    float*    otmp = (float*)(y16 + (size_t)BL * DI); // BL*DM fp32

    // ---- 0) fp32 -> f16 pre-pass (x + 4 weight matrices) ----
    auto cvt = [&](const float* s, _Float16* d, long n) {
        cvt_f32_to_f16<<<(int)((n + 255) / 256), 256, 0, stream>>>(s, d, n);
    };
    cvt(x,          x16,  (long)BL * DM);
    cvt(in_proj_w,  wi16, (long)2 * DI * DM);
    cvt(x_proj_w,   wx16, (long)96 * DI);
    cvt(dt_proj_w,  wd16, (long)DI * 64);
    cvt(out_proj_w, wo16, (long)DM * DI);

    // ---- 1) xz = x @ in_proj_w^T   [4096 x 4096], K=1024 ----
    gemm_f16_wmma<0, 1><<<dim3((2 * DI) / 64, BL / 64), 256, 0, stream>>>(
        x16, wi16, nullptr, xz16, BL, 2 * DI, DM, DM, 2 * DI);

    // ---- 2) xc = silu(conv1d(xz[:, :2048])) ----
    conv_silu_kernel<<<(BL * DI + 255) / 256, 256, 0, stream>>>(
        xz16, conv_w, conv_b, xc16, BL, L);

    // ---- 3) xdb = xc @ x_proj_w^T  [4096 x 96], K=2048 ----
    gemm_f16_wmma<0, 1><<<dim3(2, BL / 64), 256, 0, stream>>>(
        xc16, wx16, nullptr, xdb16, BL, 96, DI, DI, 96);

    // ---- 4) dt = softplus(xdb[:, :64] @ dt_proj_w^T + b)  [4096 x 2048], K=64 ----
    gemm_f16_wmma<1, 1><<<dim3(DI / 64, BL / 64), 256, 0, stream>>>(
        xdb16, wd16, dt_proj_b, dt16, BL, DI, 64, 96, DI);

    // ---- 5) selective scan -> y (includes D*xc skip + silu(z) gate) ----
    scan_kernel<<<(B * DI + 255) / 256, 256, 0, stream>>>(
        dt16, xdb16, xc16, xz16, A_log, Dp, y16, B, L);

    // ---- 6) otmp = y @ out_proj_w^T  [4096 x 1024], K=2048, fp32 out ----
    gemm_f16_wmma<0, 0><<<dim3(DM / 64, BL / 64), 256, 0, stream>>>(
        y16, wo16, nullptr, otmp, BL, DM, DI, DI, DM);

    // ---- 7) out = layernorm(otmp + x) ----
    ln_kernel<<<BL, 256, 0, stream>>>(otmp, x, ln_w, ln_b, out, DM);
}
